// NoBrainEncoderBlock_31662498906140
// MI455X (gfx1250) — matrix operand order repaired and checked by reading, compiled, not run on hardware
//
#include <hip/hip_runtime.h>
#include <math.h>

typedef __attribute__((ext_vector_type(16))) _Float16 v16h;
typedef __attribute__((ext_vector_type(8)))  float    v8f;
typedef __attribute__((ext_vector_type(4)))  float    v4f;

#define Bn 32
#define Nn 2048
#define Dn 2048

// ---------- kernel 0: ||q1[b]|| -> ws[b] ----------
__global__ __launch_bounds__(256) void qnorm_kernel(const float* __restrict__ q1,
                                                    float* __restrict__ qn) {
  const int b = blockIdx.x;
  const float* row = q1 + (size_t)b * Dn;
  float s = 0.f;
  for (int i = threadIdx.x; i < Dn; i += 256) { float v = row[i]; s = fmaf(v, v, s); }
  for (int off = 16; off > 0; off >>= 1) s += __shfl_down(s, off, 32);
  __shared__ float red[8];
  const int lane = threadIdx.x & 31, w = threadIdx.x >> 5;
  if (lane == 0) red[w] = s;
  __syncthreads();
  if (threadIdx.x == 0) {
    float t = 0.f;
#pragma unroll
    for (int i = 0; i < 8; ++i) t += red[i];
    qn[b] = sqrtf(t);
  }
}

#define PACK4(H, BASE, V)                                                      \
  do {                                                                         \
    H[(BASE) + 0] = (_Float16)(V)[0];                                          \
    H[(BASE) + 1] = (_Float16)(V)[1];                                          \
    H[(BASE) + 2] = (_Float16)(V)[2];                                          \
    H[(BASE) + 3] = (_Float16)(V)[3];                                          \
  } while (0)

// ---------- kernel 1: WMMA cosine scores (masked, clipped) ----------
// One wave handles a 16-row tile of k1 for one batch. 4096 tiles total.
__global__ __launch_bounds__(256) void scores_kernel(const float* __restrict__ q1,
                                                     const float* __restrict__ k1,
                                                     const float* __restrict__ mask,
                                                     const float* __restrict__ qn,
                                                     float* __restrict__ out) {
  const int lane = threadIdx.x & 31;
  const int wave = threadIdx.x >> 5;
  const int tile = blockIdx.x * 8 + wave;   // 0..4095
  const int b    = tile >> 7;               // tile / 128
  const int n0   = (tile & 127) << 4;       // 16-row tile base
  const int col  = lane & 15;               // which k-row this lane feeds (B-operand column)
  const int hi   = lane >> 4;               // K-half select per B layout

  // B operand: lane holds 16 contiguous d-values of row n0+col (lanes 16-31: +16 offset)
  const float* krow = k1 + (size_t)(b * Nn + n0 + col) * Dn + hi * 16;
  // A operand: q replicated across M; lanes 0-15 need K={0..15}, lanes 16-31 K={16..31}
  const float* qrow = q1 + (size_t)b * Dn + hi * 16;

  v8f acc = {0.f, 0.f, 0.f, 0.f, 0.f, 0.f, 0.f, 0.f};
  float ss = 0.f;  // per-lane partial sum of squares of this k-row

#pragma unroll 2
  for (int d0 = 0; d0 < Dn; d0 += 32) {
    __builtin_prefetch((const void*)(krow + d0 + 256), 0, 0);  // ~8 iters ahead
    // single-use stream of k1: non-temporal b128 loads
    v4f k0 = __builtin_nontemporal_load((const v4f*)(krow + d0 + 0));
    v4f kA = __builtin_nontemporal_load((const v4f*)(krow + d0 + 4));
    v4f kB = __builtin_nontemporal_load((const v4f*)(krow + d0 + 8));
    v4f kC = __builtin_nontemporal_load((const v4f*)(krow + d0 + 12));
    v4f qa = *(const v4f*)(qrow + d0 + 0);
    v4f qb = *(const v4f*)(qrow + d0 + 4);
    v4f qc = *(const v4f*)(qrow + d0 + 8);
    v4f qd = *(const v4f*)(qrow + d0 + 12);

    // ||k||^2 partial in f32 while data is in registers
    ss = fmaf(k0[0], k0[0], ss); ss = fmaf(k0[1], k0[1], ss);
    ss = fmaf(k0[2], k0[2], ss); ss = fmaf(k0[3], k0[3], ss);
    ss = fmaf(kA[0], kA[0], ss); ss = fmaf(kA[1], kA[1], ss);
    ss = fmaf(kA[2], kA[2], ss); ss = fmaf(kA[3], kA[3], ss);
    ss = fmaf(kB[0], kB[0], ss); ss = fmaf(kB[1], kB[1], ss);
    ss = fmaf(kB[2], kB[2], ss); ss = fmaf(kB[3], kB[3], ss);
    ss = fmaf(kC[0], kC[0], ss); ss = fmaf(kC[1], kC[1], ss);
    ss = fmaf(kC[2], kC[2], ss); ss = fmaf(kC[3], kC[3], ss);

    v16h Aop, Bop;
    PACK4(Aop, 0, qa);  PACK4(Aop, 4, qb);  PACK4(Aop, 8, qc);  PACK4(Aop, 12, qd);
    PACK4(Bop, 0, k0);  PACK4(Bop, 4, kA);  PACK4(Bop, 8, kB);  PACK4(Bop, 12, kC);

    // D = A(16x32 f16, q replicated) x B(32x16 f16, k-tile) + C(f32)
    acc = __builtin_amdgcn_wmma_f32_16x16x32_f16(
        /*neg_a=*/false, Aop, /*neg_b=*/false, Bop,
        /*c_mod=*/(short)0, acc, /*reuse_a=*/false, /*reuse_b=*/false);
  }

  // combine the two half-row ss partials (lane j <-> lane j+16)
  ss += __shfl_xor(ss, 16, 32);

  if (lane < 16) {
    const float dot = acc[0];            // N = lane column, all M rows identical
    const float nq  = qn[b];
    const float nk  = sqrtf(ss);
    const float qc  = fmaxf(nq, 1e-12f); // F.normalize eps
    const float kc  = fmaxf(nk, 1e-12f);
    const float ratio = (nq / qc) * (nk / kc);      // ||q1n||*||k1n||
    const float cosv  = (dot / (qc * kc)) / fmaxf(ratio, 1e-8f);
    const int n = n0 + col;
    float s = cosv * mask[b * Nn + n];
    s = fminf(fmaxf(s, 0.f), 1.f);
    out[(size_t)b * Nn + n] = s;
  }
}

// ---------- kernel 2: in-place softmax over N per batch ----------
__global__ __launch_bounds__(256) void softmax_kernel(float* __restrict__ out) {
  const int b = blockIdx.x;
  float* row = out + (size_t)b * Nn;
  const int t = threadIdx.x;
  const int lane = t & 31, w = t >> 5;

  v4f a = *(const v4f*)(row + t * 8);
  v4f c = *(const v4f*)(row + t * 8 + 4);

  float m = fmaxf(fmaxf(fmaxf(a[0], a[1]), fmaxf(a[2], a[3])),
                  fmaxf(fmaxf(c[0], c[1]), fmaxf(c[2], c[3])));
  for (int off = 16; off > 0; off >>= 1) m = fmaxf(m, __shfl_xor(m, off, 32));
  __shared__ float red[8];
  if (lane == 0) red[w] = m;
  __syncthreads();
  float bmax = red[0];
#pragma unroll
  for (int i = 1; i < 8; ++i) bmax = fmaxf(bmax, red[i]);

  v4f ea, ec;
  float s = 0.f;
#pragma unroll
  for (int i = 0; i < 4; ++i) { ea[i] = __expf(a[i] - bmax); s += ea[i]; }
#pragma unroll
  for (int i = 0; i < 4; ++i) { ec[i] = __expf(c[i] - bmax); s += ec[i]; }
  for (int off = 16; off > 0; off >>= 1) s += __shfl_xor(s, off, 32);
  __syncthreads();
  if (lane == 0) red[w] = s;
  __syncthreads();
  float tot = 0.f;
#pragma unroll
  for (int i = 0; i < 8; ++i) tot += red[i];
  const float inv = 1.f / tot;

#pragma unroll
  for (int i = 0; i < 4; ++i) ea[i] *= inv;
#pragma unroll
  for (int i = 0; i < 4; ++i) ec[i] *= inv;
  *(v4f*)(row + t * 8)     = ea;
  *(v4f*)(row + t * 8 + 4) = ec;
}

extern "C" void kernel_launch(void* const* d_in, const int* in_sizes, int n_in,
                              void* d_out, int out_size, void* d_ws, size_t ws_size,
                              hipStream_t stream) {
  // setup_inputs order: q1, k1, q2, k2, mask, temp  (q2/k2/temp unused in fwd math)
  const float* q1   = (const float*)d_in[0];
  const float* k1   = (const float*)d_in[1];
  const float* mask = (const float*)d_in[4];
  float* out = (float*)d_out;
  float* qn  = (float*)d_ws;  // 32 floats

  qnorm_kernel<<<Bn, 256, 0, stream>>>(q1, qn);
  scores_kernel<<<(Bn * (Nn / 16)) / 8, 256, 0, stream>>>(q1, k1, mask, qn, out);
  softmax_kernel<<<Bn, 256, 0, stream>>>(out);
}
